// VulMixer_55533927137752
// MI455X (gfx1250) — matrix operand order repaired and checked by reading, not compiled, not run on hardware
//
#include <hip/hip_runtime.h>
#include <hip/hip_bf16.h>

typedef __attribute__((ext_vector_type(16))) _Float16 v16h;
typedef __attribute__((ext_vector_type(8)))  _Float16 v8h;
typedef __attribute__((ext_vector_type(8)))  float    v8f;

#define HID 768
#define EMB 64
#define CPG 12            // in-channels per conv group (768/64)

// ---- LDS layout (bytes) ----
#define OFF_Q    0                       // q      : 768*64 f16 = 96 KB   (pre-scaled by 1/8)
#define OFF_K    (HID*EMB*2)             // k      : 768*64 f16 = 96 KB
#define OFF_VT   (2*HID*EMB*2)           // v^T    : 64*768 f16 = 96 KB
#define OFF_BNC  (3*HID*EMB*2)           // 16 KB: fi_w staging, later P-bounce (8 waves x 2 KB)
#define OFF_CMAX (OFF_BNC + 16384)       // 64 f32
#define OFF_CINV (OFF_CMAX + 256)        // 64 f32
#define OFF_S4   (OFF_CINV + 256)        // 4*64 f32 partials
#define LDS_BYTES (OFF_S4 + 1024)        // = 312832 B  (< 320 KB WGP limit)

__device__ __forceinline__ float gelu_exact(float x) {
    return 0.5f * x * (1.0f + erff(x * 0.70710678118654752f));
}

__device__ __forceinline__ v16h cat8(v8h lo, v8h hi) {
    return __builtin_shufflevector(lo, hi, 0,1,2,3,4,5,6,7,8,9,10,11,12,13,14,15);
}

// Butterfly reductions across the 16-lane half-wave via DS_SWIZZLE_B32
// (group-of-32 mode: offset = xor_mask<<10 | and_mask 0x1f; xor<16 never
// crosses the 16-lane boundary, so each half reduces independently).
__device__ __forceinline__ float xmax16(float v) {
    int i;
    i = __builtin_amdgcn_ds_swizzle(__float_as_int(v), 0x041f); v = fmaxf(v, __int_as_float(i));
    i = __builtin_amdgcn_ds_swizzle(__float_as_int(v), 0x081f); v = fmaxf(v, __int_as_float(i));
    i = __builtin_amdgcn_ds_swizzle(__float_as_int(v), 0x101f); v = fmaxf(v, __int_as_float(i));
    i = __builtin_amdgcn_ds_swizzle(__float_as_int(v), 0x201f); v = fmaxf(v, __int_as_float(i));
    return v;
}
__device__ __forceinline__ float xadd16(float v) {
    int i;
    i = __builtin_amdgcn_ds_swizzle(__float_as_int(v), 0x041f); v += __int_as_float(i);
    i = __builtin_amdgcn_ds_swizzle(__float_as_int(v), 0x081f); v += __int_as_float(i);
    i = __builtin_amdgcn_ds_swizzle(__float_as_int(v), 0x101f); v += __int_as_float(i);
    i = __builtin_amdgcn_ds_swizzle(__float_as_int(v), 0x201f); v += __int_as_float(i);
    return v;
}

__global__ __launch_bounds__(256)
void vulmixer_fused(const float* __restrict__ x_all,
                    const float* __restrict__ conv_w,
                    const float* __restrict__ conv_b,
                    const float* __restrict__ ln_g,
                    const float* __restrict__ ln_b,
                    const float* __restrict__ fi_w,
                    const float* __restrict__ fi_b,
                    float* __restrict__ out)
{
    extern __shared__ char smem[];
    _Float16* qbuf = (_Float16*)(smem + OFF_Q);
    _Float16* kbuf = (_Float16*)(smem + OFF_K);
    _Float16* vtb  = (_Float16*)(smem + OFF_VT);
    float*    wbuf = (float*)(smem + OFF_BNC);      // fi_w staging (preprocess phase)
    _Float16* pbnc = (_Float16*)(smem + OFF_BNC);   // P bounce (attention phase)
    float*    cmax = (float*)(smem + OFF_CMAX);
    float*    cinv = (float*)(smem + OFF_CINV);
    float*    s4   = (float*)(smem + OFF_S4);

    const int b   = blockIdx.x;
    const int tid = threadIdx.x;
    const float* __restrict__ xb = x_all + (size_t)b * HID * EMB;

    // ================= Phase A: preprocess q/k/v into LDS (f16) =================

    // stage fi_w (64x64 f32) into the bounce region
    for (int i = tid; i < EMB * EMB; i += 256) wbuf[i] = fi_w[i];

    // ---- k branch: softmax over channel axis (768) per column e, then GELU ----
    {
        const int e = tid & 63, part = tid >> 6;
        float mx = -1e30f;
        for (int c = part * 192; c < part * 192 + 192; ++c)
            mx = fmaxf(mx, xb[c * EMB + e]);
        s4[part * 64 + e] = mx;
    }
    __syncthreads();
    if (tid < 64)
        cmax[tid] = fmaxf(fmaxf(s4[tid], s4[64 + tid]), fmaxf(s4[128 + tid], s4[192 + tid]));
    __syncthreads();
    {
        const int e = tid & 63, part = tid >> 6;
        const float m = cmax[e];
        float sm = 0.f;
        for (int c = part * 192; c < part * 192 + 192; ++c)
            sm += __expf(xb[c * EMB + e] - m);
        s4[part * 64 + e] = sm;
    }
    __syncthreads();
    if (tid < 64)
        cinv[tid] = 1.0f / (s4[tid] + s4[64 + tid] + s4[128 + tid] + s4[192 + tid]);
    __syncthreads();
    for (int i = tid; i < HID * EMB; i += 256) {
        const int e = i & 63;
        const float p = __expf(xb[i] - cmax[e]) * cinv[e];
        kbuf[i] = (_Float16)gelu_exact(p);
    }

    // ---- v branch: GELU(x @ fi_w^T + fi_b), stored transposed (emb-major) ----
    for (int rr = 0; rr < 3; ++rr) {
        const int c = tid + rr * 256;
        float xv[EMB];
        const float* xr = xb + c * EMB;
        #pragma unroll
        for (int e = 0; e < EMB; ++e) xv[e] = xr[e];
        for (int j = 0; j < EMB; ++j) {
            float s = fi_b[j];
            const float* wr = wbuf + j * EMB;
            #pragma unroll
            for (int e = 0; e < EMB; ++e) s = fmaf(xv[e], wr[e], s);
            vtb[j * HID + c] = (_Float16)gelu_exact(s);
        }
    }

    // ---- q branch: grouped conv1d(k=3) + residual, GELU, LayerNorm(64) ----
    // The attention scale 1/sqrt(64)=0.125 is folded in here: exact in f16.
    for (int rr = 0; rr < 3; ++rr) {
        const int o = tid + rr * 256;
        const int g = o / CPG;
        float acc[EMB];
        const float cb = conv_b[o];
        #pragma unroll
        for (int l = 0; l < EMB; ++l) acc[l] = cb;
        const float* wrow = conv_w + o * CPG * 3;
        for (int i = 0; i < CPG; ++i) {
            const float* xr = xb + (g * CPG + i) * EMB;
            const float w0 = wrow[i * 3 + 0], w1 = wrow[i * 3 + 1], w2 = wrow[i * 3 + 2];
            #pragma unroll
            for (int l = 0; l < EMB; ++l) {
                float s = w1 * xr[l];
                if (l > 0)       s = fmaf(w0, xr[l - 1], s);
                if (l < EMB - 1) s = fmaf(w2, xr[l + 1], s);
                acc[l] += s;
            }
        }
        const float* xo = xb + o * EMB;
        float mu = 0.f;
        #pragma unroll
        for (int l = 0; l < EMB; ++l) { acc[l] = gelu_exact(acc[l] + xo[l]); mu += acc[l]; }
        mu *= (1.0f / EMB);
        float var = 0.f;
        #pragma unroll
        for (int l = 0; l < EMB; ++l) { const float d = acc[l] - mu; var = fmaf(d, d, var); }
        const float rstd = rsqrtf(var * (1.0f / EMB) + 1e-5f);
        #pragma unroll
        for (int l = 0; l < EMB; ++l)
            qbuf[o * EMB + l] =
                (_Float16)(0.125f * (((acc[l] - mu) * rstd) * ln_g[l] + ln_b[l]));
    }
    __syncthreads();

    // ================= Phase B: flash attention out = softmax(q k^T / 8) v =================
    // The reference's additive mask is constant along the softmax axis
    // (broadcast of a [768,1] tensor), so softmax(s + mask) == softmax(s): dropped.

    const int lane = tid & 31;
    const int wv   = tid >> 5;       // wave 0..7
    const int lm   = lane & 15;      // N / M index within a 16-lane half
    const int hi   = lane >> 4;      // half-wave select
    _Float16* pbuf = pbnc + wv * 16 * EMB;   // per-wave-private 16x64 f16 P tile

    for (int rc = 0; rc < 6; ++rc) {
        const int row = rc * 128 + wv * 16 + lm;

        // q A-fragments (16x32 f16): lane holds M=lm; K = kc*32 + hi*16 + 0..15 contiguous
        v16h qf[2];
        #pragma unroll
        for (int kc = 0; kc < 2; ++kc) {
            const _Float16* p0 = qbuf + row * EMB + kc * 32 + hi * 16;
            qf[kc] = cat8(*(const v8h*)p0, *(const v8h*)(p0 + 8));
        }

        float rowM[8], rowL[8];   // rowL: per-lane PARTIAL sums (reduced once at end)
        v8f oacc[4];
        #pragma unroll
        for (int r = 0; r < 8; ++r) { rowM[r] = -1e30f; rowL[r] = 0.f; }
        #pragma unroll
        for (int nt = 0; nt < 4; ++nt) oacc[nt] = {};

        for (int kt = 0; kt < 12; ++kt) {
            const int kb0 = kt * 64;

            // ---- S = (q/8) k^T : 4 N-tiles of 16 keys, K=64 in two chunks ----
            v8f s[4];
            #pragma unroll
            for (int nt = 0; nt < 4; ++nt) {
                v8f sacc = {};
                #pragma unroll
                for (int kc = 0; kc < 2; ++kc) {
                    // B (32x16): lane column n=lm (key), K = kc*32 + hi*16 + 0..15 contiguous
                    const _Float16* kp = kbuf + (size_t)(kb0 + nt * 16 + lm) * EMB + kc * 32 + hi * 16;
                    const v16h bfrag = cat8(*(const v8h*)kp, *(const v8h*)(kp + 8));
                    sacc = __builtin_amdgcn_wmma_f32_16x16x32_f16(
                        false, qf[kc], false, bfrag, (short)0, sacc, false, false);
                }
                s[nt] = sacc;
            }

            // ---- online softmax: lane-replicated row max, per-lane partial sums ----
            #pragma unroll
            for (int r = 0; r < 8; ++r) {
                float t = fmaxf(fmaxf(s[0][r], s[1][r]), fmaxf(s[2][r], s[3][r]));
                t = xmax16(t);
                const float mNew  = fmaxf(rowM[r], t);
                const float alpha = __expf(rowM[r] - mNew);   // group-uniform
                rowM[r] = mNew;
                float rs = 0.f;
                #pragma unroll
                for (int nt = 0; nt < 4; ++nt) {
                    const float p = __expf(s[nt][r] - mNew);
                    s[nt][r] = p;
                    rs += p;
                }
                rowL[r] = rowL[r] * alpha + rs;               // partial, consistent scaling
                #pragma unroll
                for (int nt = 0; nt < 4; ++nt) oacc[nt][r] *= alpha;
            }

            // ---- bounce P through per-wave LDS (C-layout -> A-layout) ----
            #pragma unroll
            for (int nt = 0; nt < 4; ++nt)
                #pragma unroll
                for (int r = 0; r < 8; ++r)
                    pbuf[(r + 8 * hi) * EMB + nt * 16 + lm] = (_Float16)s[nt][r];
            // Intra-wave visibility: LDS ops of one wave complete in order;
            // after DScnt==0 all lanes' stores are readable by all lanes.
            asm volatile("s_wait_dscnt 0" ::: "memory");

            v16h pf[2];
            #pragma unroll
            for (int kc = 0; kc < 2; ++kc) {
                const _Float16* pp = pbuf + lm * EMB + kc * 32 + hi * 16;
                pf[kc] = cat8(*(const v8h*)pp, *(const v8h*)(pp + 8));
            }

            // ---- O += P v : B from v^T, column n=emb, K=key contiguous ----
            #pragma unroll
            for (int nt = 0; nt < 4; ++nt) {
                #pragma unroll
                for (int kc = 0; kc < 2; ++kc) {
                    const _Float16* vp = vtb + (size_t)(nt * 16 + lm) * HID + kb0 + kc * 32 + hi * 16;
                    const v16h bfrag = cat8(*(const v8h*)vp, *(const v8h*)(vp + 8));
                    oacc[nt] = __builtin_amdgcn_wmma_f32_16x16x32_f16(
                        false, pf[kc], false, bfrag, (short)0, oacc[nt], false, false);
                }
            }
        }

        // ---- reduce the deferred row sums once, normalize, store ----
        float* ob = out + ((size_t)b * HID + rc * 128 + wv * 16) * EMB;
        #pragma unroll
        for (int r = 0; r < 8; ++r) {
            const float inv = 1.0f / xadd16(rowL[r]);
            #pragma unroll
            for (int nt = 0; nt < 4; ++nt)
                ob[(r + 8 * hi) * EMB + nt * 16 + lm] = oacc[nt][r] * inv;
        }
    }
}

extern "C" void kernel_launch(void* const* d_in, const int* in_sizes, int n_in,
                              void* d_out, int out_size, void* d_ws, size_t ws_size,
                              hipStream_t stream) {
    const float* x      = (const float*)d_in[0];
    const float* conv_w = (const float*)d_in[1];
    const float* conv_b = (const float*)d_in[2];
    const float* ln_g   = (const float*)d_in[3];
    const float* ln_b   = (const float*)d_in[4];
    const float* fi_w   = (const float*)d_in[5];
    const float* fi_b   = (const float*)d_in[6];
    // d_in[7] (bi_w) / d_in[8] (bi_b) unused: the additive mask is constant
    // along the softmax axis, so it cancels exactly in softmax.
    const int bn = in_sizes[0] / (HID * EMB);   // 128 batches
    vulmixer_fused<<<dim3(bn), dim3(256), LDS_BYTES, stream>>>(
        x, conv_w, conv_b, ln_g, ln_b, fi_w, fi_b, (float*)d_out);
    (void)n_in; (void)out_size; (void)d_ws; (void)ws_size;
}
